// PhotoVoModel_34617436405934
// MI455X (gfx1250) — compile-verified, matrix operand order checked by reading, not run on hardware
//
#include <hip/hip_runtime.h>
#include <hip/hip_bf16.h>

typedef __attribute__((ext_vector_type(16))) _Float16 v16h;
typedef __attribute__((ext_vector_type(8)))  _Float16 v8h;
typedef __attribute__((ext_vector_type(8)))  float    v8f;

#define PD    768      // 3*16*16 patch dim
#define NKP   2048     // keypoints per image
#define NM    1024     // matches
#define TWOM  2048
#define BB    32
#define DD    1024
#define NOUT  1023
#define HH    256
#define WW    256
#define ROWS_PER_WG 64 // 4 WMMA M-tiles per wave
#define APAD  776      // 768 + 8 halfs pad -> LDS row stride 388 dwords (388%64=4, conflict-free)

// dynamic LDS layout: As[64][APAD] f16, then f32/i32 row-meta arrays
#define SMEM_AS_BYTES (ROWS_PER_WG * APAD * 2)
#define SMEM_BYTES    (SMEM_AS_BYTES + ROWS_PER_WG * 4 * 8)

static __device__ inline v16h mk16(v8h lo, v8h hi) {
  return __builtin_shufflevector(lo, hi, 0,1,2,3,4,5,6,7,8,9,10,11,12,13,14,15);
}

// ---------------------------------------------------------------------------
// Convert W_lin (768 x 1023 f32) into f16 Wt[n][k] (1024 x 768), transposed so
// B-fragments read contiguous K; padded column 1023 is zero (becomes score slot).
// ---------------------------------------------------------------------------
__global__ void k_convert_w(const float* __restrict__ W, _Float16* __restrict__ Wt) {
  int t = blockIdx.x * blockDim.x + threadIdx.x;
  if (t >= DD * PD) return;
  int n = t / PD, k = t - n * PD;
  Wt[(size_t)n * PD + k] = (n < NOUT) ? (_Float16)W[(size_t)k * NOUT + n] : (_Float16)0.0f;
}

// ---------------------------------------------------------------------------
// kpts0_valid / kpts1_valid outputs
// ---------------------------------------------------------------------------
__global__ void k_kpts(const float* __restrict__ kpts0, const float* __restrict__ kpts1,
                       const long long* __restrict__ matches,
                       float* __restrict__ ok0, float* __restrict__ ok1) {
  int i = blockIdx.x * blockDim.x + threadIdx.x;
  if (i >= BB * NM) return;
  int b = i / NM;
  long long i0 = matches[(size_t)i * 2 + 0];
  long long i1 = matches[(size_t)i * 2 + 1];
  bool mask = i1 > -1;
  float nanv = __builtin_nanf("");
  long long g0 = mask ? i0 : 0, g1 = mask ? i1 : 0;
  ok0[(size_t)i * 2 + 0] = mask ? kpts0[((size_t)b * NKP + g0) * 2 + 0] : nanv;
  ok0[(size_t)i * 2 + 1] = mask ? kpts0[((size_t)b * NKP + g0) * 2 + 1] : nanv;
  ok1[(size_t)i * 2 + 0] = mask ? kpts1[((size_t)b * NKP + g1) * 2 + 0] : nanv;
  ok1[(size_t)i * 2 + 1] = mask ? kpts1[((size_t)b * NKP + g1) * 2 + 1] : nanv;
}

// ---------------------------------------------------------------------------
// pred_vo head: pooled = mean(image_embs, -1);  pred = pooled @ fc_W + fc_b
// ---------------------------------------------------------------------------
__global__ void k_predvo(const float* __restrict__ embs, const float* __restrict__ fcW,
                         const float* __restrict__ fcb, float* __restrict__ out) {
  __shared__ float pooled[DD];
  int b = blockIdx.x;
  for (int d = threadIdx.x; d < DD; d += blockDim.x) {
    const float* p = embs + ((size_t)b * DD + d) * 64;
    float s = 0.0f;
    #pragma unroll 8
    for (int j = 0; j < 64; ++j) s += p[j];
    pooled[d] = s * (1.0f / 64.0f);
  }
  __syncthreads();
  if (threadIdx.x < 6) {
    int j = threadIdx.x;
    float a = fcb[j];
    for (int d = 0; d < DD; ++d) a += pooled[d] * fcW[d * 6 + j];
    out[b * 6 + j] = a;
  }
}

// ---------------------------------------------------------------------------
// Fused: patch gather -> LN1 -> WMMA GEMM (f16 in / f32 acc) -> +bias -> LN2
//        -> concat score -> patch_embs      (64 rows/WG, wave w owns 128 N-cols)
// ---------------------------------------------------------------------------
__global__ __launch_bounds__(256)
void k_main(const float* __restrict__ image0, const float* __restrict__ image1,
            const float* __restrict__ kpts0,  const float* __restrict__ kpts1,
            const float* __restrict__ scores0,const float* __restrict__ scores1,
            const long long* __restrict__ matches,
            const float* __restrict__ ln1_g,  const float* __restrict__ ln1_b,
            const float* __restrict__ b_lin,
            const float* __restrict__ ln2_g,  const float* __restrict__ ln2_b,
            const _Float16* __restrict__ Wt,
            float* __restrict__ out_pe) {
  extern __shared__ char smem[];
  _Float16 (*As)[APAD] = (_Float16 (*)[APAD])smem;
  float* sScore   = (float*)(smem + SMEM_AS_BYTES);
  float* rowsum   = sScore   + ROWS_PER_WG;
  float* rowsumsq = rowsum   + ROWS_PER_WG;
  int*   sX    = (int*)(rowsumsq + ROWS_PER_WG);
  int*   sY    = sX    + ROWS_PER_WG;
  int*   sMask = sY    + ROWS_PER_WG;
  int*   sImg  = sMask + ROWS_PER_WG;
  int*   sBt   = sImg  + ROWS_PER_WG;

  const int tid  = threadIdx.x;
  const int lane = tid & 31;
  const int wv   = tid >> 5;
  const size_t row0 = (size_t)blockIdx.x * ROWS_PER_WG;

  // ---- phase 0: per-row metadata (one thread per row) ----
  if (tid < ROWS_PER_WG) {
    size_t grow = row0 + tid;
    int b  = (int)(grow / TWOM);
    int mm = (int)(grow % TWOM);
    int is1 = mm >= NM;
    int m   = is1 ? mm - NM : mm;
    long long i0 = matches[((size_t)b * NM + m) * 2 + 0];
    long long i1 = matches[((size_t)b * NM + m) * 2 + 1];
    int mask = (i1 > -1);
    long long idx = mask ? (is1 ? i1 : i0) : 0;
    const float* kp = is1 ? kpts1 : kpts0;
    const float* sc = is1 ? scores1 : scores0;
    float kx = mask ? kp[((size_t)b * NKP + idx) * 2 + 0] : 0.0f;
    float ky = mask ? kp[((size_t)b * NKP + idx) * 2 + 1] : 0.0f;
    sX[tid] = (int)rintf(kx);
    sY[tid] = (int)rintf(ky);
    sMask[tid] = mask;
    sImg[tid]  = is1;
    sBt[tid]   = b;
    sScore[tid] = mask ? sc[(size_t)b * NKP + idx] : 0.0f;
    rowsum[tid] = 0.0f;
    rowsumsq[tid] = 0.0f;
  }
  __syncthreads();

  // ---- phase 1: gather patch + LN1 -> f16 LDS A-tile (wave w: rows 8w..8w+7) ----
  for (int i = 0; i < 8; ++i) {
    int r   = wv * 8 + i;
    int b   = sBt[r], is1 = sImg[r], msk = sMask[r];
    int x0  = sX[r],  y0  = sY[r];
    const float* img = is1 ? image1 : image0;
    float v[24];
    float s = 0.0f, s2 = 0.0f;
    #pragma unroll
    for (int j = 0; j < 24; ++j) {
      int e = j * 32 + lane;                 // e = c*256 + py*16 + px
      float val;
      if (msk) {                             // mask is row-uniform -> no divergence
        int c  = e >> 8;
        int py = (e >> 4) & 15;
        int px = e & 15;
        int yy = min(max(y0 + py - 8, 0), HH - 1);
        int xx = min(max(x0 + px - 8, 0), WW - 1);
        val = img[(((size_t)b * 3 + c) * HH + yy) * WW + xx];
      } else {
        val = -1.0f;
      }
      v[j] = val; s += val; s2 += val * val;
    }
    #pragma unroll
    for (int o = 16; o > 0; o >>= 1) {
      s  += __shfl_xor(s,  o, 32);
      s2 += __shfl_xor(s2, o, 32);
    }
    float mu  = s * (1.0f / PD);
    float var = s2 * (1.0f / PD) - mu * mu;
    float rst = rsqrtf(var + 1e-5f);
    #pragma unroll
    for (int j = 0; j < 24; ++j) {
      int e = j * 32 + lane;
      float xh = (v[j] - mu) * rst * ln1_g[e] + ln1_b[e];
      As[r][e] = (_Float16)xh;
    }
  }
  __syncthreads();

  // ---- phase 2: WMMA GEMM. 4 M-tiles x 8 N-tiles; whole K-slice of B batched
  //      into registers, double-buffered across K so waits are partial, not 0 ----
  const int hi    = lane >> 4;
  const int l15   = lane & 15;
  const int nbase = wv * 128;

  v8f zacc = {};
  v8f acc[4][8];
  #pragma unroll
  for (int mt = 0; mt < 4; ++mt)
    #pragma unroll
    for (int nt = 0; nt < 8; ++nt) acc[mt][nt] = zacc;

  const _Float16* wbase = Wt + (size_t)(nbase + l15) * PD;

  auto loadB = [&](int kc, v8h buf[8][2]) {
    const int ko0 = kc + 8 * hi;            // K layout: VGPR0-3 hold K=[8*hi,8*hi+8)
    const int ko1 = ko0 + 16;               //           VGPR4-7 hold K=[16+8*hi,..)
    #pragma unroll
    for (int nt = 0; nt < 8; ++nt) {
      const _Float16* wn = wbase + (size_t)nt * 16 * PD;
      buf[nt][0] = *(const v8h*)(wn + ko0);
      buf[nt][1] = *(const v8h*)(wn + ko1);
    }
  };
  auto compute = [&](int kc, v8h buf[8][2]) {
    const int ko0 = kc + 8 * hi;
    const int ko1 = ko0 + 16;
    v16h a[4];
    #pragma unroll
    for (int mt = 0; mt < 4; ++mt)
      a[mt] = mk16(*(const v8h*)&As[mt * 16 + l15][ko0],
                   *(const v8h*)&As[mt * 16 + l15][ko1]);
    #pragma unroll
    for (int nt = 0; nt < 8; ++nt) {
      v16h bm = mk16(buf[nt][0], buf[nt][1]);
      #pragma unroll
      for (int mt = 0; mt < 4; ++mt)
        acc[mt][nt] = __builtin_amdgcn_wmma_f32_16x16x32_f16(
            false, a[mt], false, bm, (short)0, acc[mt][nt], false, false);
    }
  };

  v8h b0[8][2], b1[8][2];
  loadB(0, b0);                              // prologue: slice 0 in flight
  for (int kc = 0; kc < PD; kc += 64) {      // two K-slices per iteration
    loadB(kc + 32, b1);                      // issue slice kc+32 loads
    if (kc + 64 < PD)
      __builtin_prefetch(wbase + kc + 64, 0, 3);
    compute(kc, b0);                         // WMMAs hide b1's L2 latency
    if (kc + 64 < PD)
      loadB(kc + 64, b0);                    // issue slice kc+64 loads
    compute(kc + 32, b1);                    // WMMAs hide b0's L2 latency
  }

  // ---- phase 3: +bias, LN2 row stats (padded col 1023 contributes exactly 0) ----
  #pragma unroll
  for (int nt = 0; nt < 8; ++nt) {
    int n = nbase + nt * 16 + l15;
    float bias = (n < NOUT) ? b_lin[n] : 0.0f;
    #pragma unroll
    for (int mt = 0; mt < 4; ++mt)
      #pragma unroll
      for (int vv = 0; vv < 8; ++vv) acc[mt][nt][vv] += bias;
  }
  #pragma unroll
  for (int mt = 0; mt < 4; ++mt) {
    #pragma unroll
    for (int vv = 0; vv < 8; ++vv) {
      float ps = 0.0f, ps2 = 0.0f;
      #pragma unroll
      for (int nt = 0; nt < 8; ++nt) { float x = acc[mt][nt][vv]; ps += x; ps2 += x * x; }
      #pragma unroll
      for (int o = 8; o > 0; o >>= 1) {      // reduce the 16 lanes sharing this row
        ps  += __shfl_xor(ps,  o, 32);
        ps2 += __shfl_xor(ps2, o, 32);
      }
      int r = mt * 16 + 8 * hi + vv;         // C layout: lane group hi, VGPR vv -> row
      if (l15 == 0) { atomicAdd(&rowsum[r], ps); atomicAdd(&rowsumsq[r], ps2); }
    }
  }
  __syncthreads();

  // ---- write: LN2 + fused score in column 1023 (streaming stores) ----
  #pragma unroll
  for (int mt = 0; mt < 4; ++mt) {
    #pragma unroll
    for (int vv = 0; vv < 8; ++vv) {
      int r = mt * 16 + 8 * hi + vv;
      float mu  = rowsum[r]   * (1.0f / NOUT);
      float var = rowsumsq[r] * (1.0f / NOUT) - mu * mu;
      float rst = rsqrtf(var + 1e-5f);
      float* op = out_pe + (row0 + r) * DD;
      #pragma unroll
      for (int nt = 0; nt < 8; ++nt) {
        int n = nbase + nt * 16 + l15;
        float x = acc[mt][nt][vv];
        float yv = (n < NOUT) ? ((x - mu) * rst * ln2_g[n] + ln2_b[n]) : sScore[r];
        __builtin_nontemporal_store(yv, &op[n]);
      }
    }
  }
}

// ---------------------------------------------------------------------------
extern "C" void kernel_launch(void* const* d_in, const int* in_sizes, int n_in,
                              void* d_out, int out_size, void* d_ws, size_t ws_size,
                              hipStream_t stream) {
  (void)in_sizes; (void)n_in; (void)out_size; (void)ws_size;
  const float*     image0    = (const float*)d_in[0];
  const float*     image1    = (const float*)d_in[1];
  const float*     kpts0     = (const float*)d_in[2];
  const float*     kpts1     = (const float*)d_in[3];
  const float*     scores0   = (const float*)d_in[4];
  const float*     scores1   = (const float*)d_in[5];
  const long long* matches   = (const long long*)d_in[6];   // int64 per reference
  const float*     image_embs= (const float*)d_in[7];
  const float*     ln1_g     = (const float*)d_in[8];
  const float*     ln1_b     = (const float*)d_in[9];
  const float*     W_lin     = (const float*)d_in[10];
  const float*     b_lin     = (const float*)d_in[11];
  const float*     ln2_g     = (const float*)d_in[12];
  const float*     ln2_b     = (const float*)d_in[13];
  const float*     fc_W      = (const float*)d_in[14];
  const float*     fc_b      = (const float*)d_in[15];

  float* out    = (float*)d_out;
  float* out_pv = out;                                         // 32*6
  float* out_pe = out + 192;                                   // 32*2048*1024
  float* out_k0 = out_pe + (size_t)BB * TWOM * DD;             // 32*1024*2
  float* out_k1 = out_k0 + (size_t)BB * NM * 2;                // 32*1024*2

  _Float16* Wt = (_Float16*)d_ws;                              // 1024*768*2 = 1.5 MB

  k_convert_w<<<(DD * PD + 255) / 256, 256, 0, stream>>>(W_lin, Wt);
  k_kpts<<<(BB * NM + 255) / 256, 256, 0, stream>>>(kpts0, kpts1, matches, out_k0, out_k1);
  k_predvo<<<BB, 256, 0, stream>>>(image_embs, fc_W, fc_b, out_pv);
  k_main<<<(BB * TWOM) / ROWS_PER_WG, 256, SMEM_BYTES, stream>>>(
      image0, image1, kpts0, kpts1, scores0, scores1, matches,
      ln1_g, ln1_b, b_lin, ln2_g, ln2_b, Wt, out_pe);
}
